// energy_latencyGNN_10_1_41446434406430
// MI455X (gfx1250) — matrix operand order repaired and checked by reading, not compiled, not run on hardware
//
#include <hip/hip_runtime.h>
#include <math.h>

typedef _Float16 half_t;
typedef __attribute__((ext_vector_type(16))) _Float16 v16h;
typedef __attribute__((ext_vector_type(8)))  _Float16 v8h;
typedef __attribute__((ext_vector_type(8)))  float    v8f;

#define BB 32768
#define NN 20
#define EE 100
#define EMBD 5
#define SLOPE 0.01f

#define XCAT_W 224   // 100 gnn + 120 aux + 4 zero pad (halves)

// workspace layout, in units of half_t
#define W1S_OFF 0        // 7*8*512  = 28672
#define W2S_OFF 28672    // 4*8*512  = 16384
#define W3S_OFF 45056    // 4*4*512  =  8192
#define W4S_OFF 53248    // 2*1*512  =  1024
#define WTOT    54272    // total swizzled weight halves (108,544 B)
#define XCAT_OFF 54272   // 32768*224 halves

__device__ __forceinline__ float leaky(float x) { return x >= 0.f ? x : SLOPE * x; }
__device__ __forceinline__ float sigm(float x)  { return 1.f / (1.f + __expf(-x)); }

__device__ __forceinline__ v16h mk16(v8h lo, v8h hi) {
  union { v16h v; v8h h[2]; } u;
  u.h[0] = lo; u.h[1] = hi;
  return u.v;
}

// ---------------------------------------------------------------------------
// Stage 0: fp32 weights -> f16, pre-swizzled into WMMA B-fragment order.
// Fragment (kc,nc) covers K rows [kc*32, kc*32+32), N cols [nc*16, nc*16+16).
// Within a fragment of 512 halves: lane L owns 16 contiguous halves, where
//   N = nc*16 + (L & 15),  K = kc*32 + ((L >= 16) ? 8 : 0) + (h < 8 ? h : h + 8),  h = 0..15
// matching the ISA 16-bit B-matrix (32x16) wave32 layout.
// ---------------------------------------------------------------------------
__global__ void wswz_kernel(const float* __restrict__ W, int Kreal, int Nreal,
                            int Kc, int Nc, half_t* __restrict__ out) {
  int tid = blockIdx.x * blockDim.x + threadIdx.x;
  int total = Kc * Nc * 512;
  if (tid >= total) return;
  int frag = tid >> 9;
  int w    = tid & 511;
  int lane = w >> 4;
  int h    = w & 15;
  int kc = frag / Nc;
  int nc = frag - kc * Nc;
  int k = kc * 32 + ((lane >> 4) << 3) + (h < 8 ? h : h + 8);
  int n = nc * 16 + (lane & 15);
  float v = (k < Kreal && n < Nreal) ? W[k * Nreal + n] : 0.0f;
  out[tid] = (half_t)v;
}

// ---------------------------------------------------------------------------
// Stage 1: tiny gated GNN, one thread per sample. Node state lives in LDS so
// dynamic node indexing (h[src[e]]) stays on ds_* instead of scratch.
// Writes f16 activation row [x(100) | d(120) | 0 pad(4)] into xcat.
// ---------------------------------------------------------------------------
__global__ void __launch_bounds__(64) gnn_kernel(
    const float* __restrict__ data, const float* __restrict__ dmat,
    const int* __restrict__ src, const int* __restrict__ dst,
    const float* __restrict__ g0U, const float* __restrict__ g0V,
    const float* __restrict__ g0A, const float* __restrict__ g0B,
    const float* __restrict__ g0bU, const float* __restrict__ g0bV,
    const float* __restrict__ g0bA, const float* __restrict__ g0bB,
    const float* __restrict__ gU, const float* __restrict__ gV,
    const float* __restrict__ gA, const float* __restrict__ gB,
    const float* __restrict__ gbU, const float* __restrict__ gbV,
    const float* __restrict__ gbA, const float* __restrict__ gbB,
    half_t* __restrict__ xcat) {
  __shared__ float hs[64 * NN * EMBD];   // current node embeddings
  __shared__ float ag[64 * NN * EMBD];   // aggregation buffer
  __shared__ float h0s[64 * NN];         // layer-0 scalar inputs

  const int tid = threadIdx.x;
  const int b   = blockIdx.x * 64 + tid;
  float* h = hs + tid * (NN * EMBD);
  float* a = ag + tid * (NN * EMBD);
  float* h0 = h0s + tid * NN;

  // ---- layer 0 (input dim 1) ----
  #pragma unroll
  for (int n = 0; n < NN; ++n) h0[n] = data[b * NN + n];
  #pragma unroll
  for (int i = 0; i < NN * EMBD; ++i) a[i] = 0.f;

  float U0[EMBD], V0[EMBD], A0w[EMBD], B0w[EMBD];
  float bU0[EMBD], bV0[EMBD], bA0[EMBD], bB0[EMBD];
  #pragma unroll
  for (int c = 0; c < EMBD; ++c) {
    U0[c] = g0U[c];  V0[c] = g0V[c];  A0w[c] = g0A[c];  B0w[c] = g0B[c];
    bU0[c] = g0bU[c]; bV0[c] = g0bV[c]; bA0[c] = g0bA[c]; bB0[c] = g0bB[c];
  }
  for (int e = 0; e < EE; ++e) {
    int s = src[e], t = dst[e];
    float hu = h0[s], hv = h0[t];
    #pragma unroll
    for (int c = 0; c < EMBD; ++c) {
      float g = sigm(hu * A0w[c] + bA0[c] + hv * B0w[c] + bB0[c]);
      a[t * EMBD + c] += g * (hu * V0[c] + bV0[c]);
    }
  }
  #pragma unroll
  for (int n = 0; n < NN; ++n)
    #pragma unroll
    for (int c = 0; c < EMBD; ++c)
      h[n * EMBD + c] = leaky(h0[n] * U0[c] + bU0[c] + a[n * EMBD + c]);

  // ---- layers 1..2 (EMBD x EMBD) ----
  for (int l = 0; l < 2; ++l) {
    float U[25], V[25], Am[25], Bm[25], bU[5], bV[5], bA[5], bB[5];
    #pragma unroll
    for (int i = 0; i < 25; ++i) {
      U[i]  = gU[l * 25 + i];  V[i]  = gV[l * 25 + i];
      Am[i] = gA[l * 25 + i];  Bm[i] = gB[l * 25 + i];
    }
    #pragma unroll
    for (int c = 0; c < EMBD; ++c) {
      bU[c] = gbU[l * EMBD + c]; bV[c] = gbV[l * EMBD + c];
      bA[c] = gbA[l * EMBD + c]; bB[c] = gbB[l * EMBD + c];
    }
    #pragma unroll
    for (int i = 0; i < NN * EMBD; ++i) a[i] = 0.f;

    for (int e = 0; e < EE; ++e) {
      int s = src[e], t = dst[e];
      float hu[EMBD], hv[EMBD];
      #pragma unroll
      for (int k = 0; k < EMBD; ++k) { hu[k] = h[s * EMBD + k]; hv[k] = h[t * EMBD + k]; }
      #pragma unroll
      for (int c = 0; c < EMBD; ++c) {
        float z  = bA[c] + bB[c];
        float zv = bV[c];
        #pragma unroll
        for (int k = 0; k < EMBD; ++k) {
          z  += hu[k] * Am[k * EMBD + c] + hv[k] * Bm[k * EMBD + c];
          zv += hu[k] * V[k * EMBD + c];
        }
        a[t * EMBD + c] += sigm(z) * zv;
      }
    }
    #pragma unroll
    for (int n = 0; n < NN; ++n) {
      float tmp[EMBD];
      #pragma unroll
      for (int c = 0; c < EMBD; ++c) {
        float z = bU[c] + a[n * EMBD + c];
        #pragma unroll
        for (int k = 0; k < EMBD; ++k) z += h[n * EMBD + k] * U[k * EMBD + c];
        tmp[c] = leaky(z);
      }
      #pragma unroll
      for (int c = 0; c < EMBD; ++c) h[n * EMBD + c] = tmp[c];
    }
  }

  // ---- emit f16 row [x | d | pad] ----
  half_t* xr = xcat + (size_t)b * XCAT_W;
  #pragma unroll
  for (int i = 0; i < NN * EMBD; ++i) xr[i] = (half_t)h[i];
  for (int j = 0; j < 120; ++j) xr[100 + j] = (half_t)dmat[b * 120 + j];
  #pragma unroll
  for (int j = 0; j < 4; ++j) xr[220 + j] = (half_t)0.f;
}

// ---------------------------------------------------------------------------
// Stage 2: WMMA MLP. One wave per 16-row tile, 8 waves per block.
// All swizzled weights (108.5 KB) are staged ONCE per workgroup into LDS
// (CDNA5: 320 KB/WGP) and shared by the block's 8 waves -> global/L2 weight
// traffic drops 8x vs per-wave re-reads; B-fragments come from ds_load_b128.
//   L1: 16x224 @ 224x128   (7 K-chunks x 8 N-tiles)
//   L2: 16x128 @ 128x128   (4 x 8)
//   L3: 16x128 @ 128x64    (4 x 4)
//   L4: 16x64  @ 64x16pad  (2 x 1), sigmoid, store cols 0..1
// Inter-layer activation re-layout (C/D layout -> A layout) via per-wave
// LDS ping-pong buffers.
// ---------------------------------------------------------------------------
__global__ void __launch_bounds__(256) mlp_kernel(
    const half_t* __restrict__ wsH,
    const float* __restrict__ b1, const float* __restrict__ b2,
    const float* __restrict__ b3, const float* __restrict__ b4,
    float* __restrict__ out) {
  __shared__ __align__(32) half_t sW[WTOT];        // 108,544 B: all weights
  __shared__ __align__(32) half_t sA[8][16 * 128]; // 32 KB ping
  __shared__ __align__(32) half_t sB[8][16 * 128]; // 32 KB pong

  const int tid   = threadIdx.x;
  const int lane  = tid & 31;
  const int wslot = tid >> 5;
  const int wave  = blockIdx.x * 8 + wslot;
  const int r0    = wave * 16;

  // ---- cooperative weight staging: global(L2) -> LDS, b128 granules ----
  {
    const v8h* src = (const v8h*)wsH;
    v8h* dstp = (v8h*)sW;
    for (int i = tid; i < (WTOT / 8); i += 256) dstp[i] = src[i];
  }
  __syncthreads();

  const half_t* w1s = sW + W1S_OFF;
  const half_t* w2s = sW + W2S_OFF;
  const half_t* w3s = sW + W3S_OFF;
  const half_t* w4s = sW + W4S_OFF;
  const half_t* xrow = wsH + XCAT_OFF + (size_t)r0 * XCAT_W;

  const int m  = lane & 15;         // A-layout: row; C/D-layout: column
  const int hb = (lane >> 4) * 8;   // A-layout K sub-offset
  const int mb = (lane >> 4) * 8;   // C/D-layout row base

  half_t* bufA = sA[wslot];
  half_t* bufB = sB[wslot];

  // ---- layer 1: A-frags from global xcat, B-frags from LDS ----
  #pragma unroll
  for (int j = 0; j < 8; ++j) {
    v8f c = {};
    #pragma unroll
    for (int kc = 0; kc < 7; ++kc) {
      const int k0 = kc * 32;
      v8h a0 = *(const v8h*)(xrow + m * XCAT_W + k0 + hb);
      v8h a1 = *(const v8h*)(xrow + m * XCAT_W + k0 + 16 + hb);
      const half_t* bp = w1s + (kc * 8 + j) * 512 + lane * 16;
      v8h bb0 = *(const v8h*)(bp);
      v8h bb1 = *(const v8h*)(bp + 8);
      c = __builtin_amdgcn_wmma_f32_16x16x32_f16(false, mk16(a0, a1),
                                                 false, mk16(bb0, bb1),
                                                 (short)0, c, false, false);
    }
    const int n = j * 16 + m;
    const float bias = b1[n];
    #pragma unroll
    for (int r = 0; r < 8; ++r) {
      float v = leaky(c[r] + bias);
      bufA[(mb + r) * 128 + n] = (half_t)v;
    }
  }
  asm volatile("s_wait_dscnt 0" ::: "memory");

  // ---- layer 2 ----
  #pragma unroll
  for (int j = 0; j < 8; ++j) {
    v8f c = {};
    #pragma unroll
    for (int kc = 0; kc < 4; ++kc) {
      const int k0 = kc * 32;
      v8h a0 = *(const v8h*)(bufA + m * 128 + k0 + hb);
      v8h a1 = *(const v8h*)(bufA + m * 128 + k0 + 16 + hb);
      const half_t* bp = w2s + (kc * 8 + j) * 512 + lane * 16;
      v8h bb0 = *(const v8h*)(bp);
      v8h bb1 = *(const v8h*)(bp + 8);
      c = __builtin_amdgcn_wmma_f32_16x16x32_f16(false, mk16(a0, a1),
                                                 false, mk16(bb0, bb1),
                                                 (short)0, c, false, false);
    }
    const int n = j * 16 + m;
    const float bias = b2[n];
    #pragma unroll
    for (int r = 0; r < 8; ++r) {
      float v = leaky(c[r] + bias);
      bufB[(mb + r) * 128 + n] = (half_t)v;
    }
  }
  asm volatile("s_wait_dscnt 0" ::: "memory");

  // ---- layer 3 ----
  #pragma unroll
  for (int j = 0; j < 4; ++j) {
    v8f c = {};
    #pragma unroll
    for (int kc = 0; kc < 4; ++kc) {
      const int k0 = kc * 32;
      v8h a0 = *(const v8h*)(bufB + m * 128 + k0 + hb);
      v8h a1 = *(const v8h*)(bufB + m * 128 + k0 + 16 + hb);
      const half_t* bp = w3s + (kc * 4 + j) * 512 + lane * 16;
      v8h bb0 = *(const v8h*)(bp);
      v8h bb1 = *(const v8h*)(bp + 8);
      c = __builtin_amdgcn_wmma_f32_16x16x32_f16(false, mk16(a0, a1),
                                                 false, mk16(bb0, bb1),
                                                 (short)0, c, false, false);
    }
    const int n = j * 16 + m;
    const float bias = b3[n];
    #pragma unroll
    for (int r = 0; r < 8; ++r) {
      float v = leaky(c[r] + bias);
      bufA[(mb + r) * 64 + n] = (half_t)v;   // width 64 now
    }
  }
  asm volatile("s_wait_dscnt 0" ::: "memory");

  // ---- layer 4: 64 -> 2 (N padded to 16), sigmoid, store ----
  {
    v8f c = {};
    #pragma unroll
    for (int kc = 0; kc < 2; ++kc) {
      const int k0 = kc * 32;
      v8h a0 = *(const v8h*)(bufA + m * 64 + k0 + hb);
      v8h a1 = *(const v8h*)(bufA + m * 64 + k0 + 16 + hb);
      const half_t* bp = w4s + kc * 512 + lane * 16;
      v8h bb0 = *(const v8h*)(bp);
      v8h bb1 = *(const v8h*)(bp + 8);
      c = __builtin_amdgcn_wmma_f32_16x16x32_f16(false, mk16(a0, a1),
                                                 false, mk16(bb0, bb1),
                                                 (short)0, c, false, false);
    }
    const int n = m;
    if (n < 2) {
      const float bias = b4[n];
      #pragma unroll
      for (int r = 0; r < 8; ++r) {
        float v = sigm(c[r] + bias);
        out[(size_t)(r0 + mb + r) * 2 + n] = v;
      }
    }
  }
}

// ---------------------------------------------------------------------------
extern "C" void kernel_launch(void* const* d_in, const int* in_sizes, int n_in,
                              void* d_out, int out_size, void* d_ws, size_t ws_size,
                              hipStream_t stream) {
  const float* data = (const float*)d_in[0];
  const float* dmat = (const float*)d_in[1];
  const int*   src  = (const int*)d_in[2];
  const int*   dst  = (const int*)d_in[3];
  const float* g0U  = (const float*)d_in[4];
  const float* g0V  = (const float*)d_in[5];
  const float* g0A  = (const float*)d_in[6];
  const float* g0B  = (const float*)d_in[7];
  const float* g0bU = (const float*)d_in[8];
  const float* g0bV = (const float*)d_in[9];
  const float* g0bA = (const float*)d_in[10];
  const float* g0bB = (const float*)d_in[11];
  const float* gU   = (const float*)d_in[12];
  const float* gV   = (const float*)d_in[13];
  const float* gA   = (const float*)d_in[14];
  const float* gB   = (const float*)d_in[15];
  const float* gbU  = (const float*)d_in[16];
  const float* gbV  = (const float*)d_in[17];
  const float* gbA  = (const float*)d_in[18];
  const float* gbB  = (const float*)d_in[19];
  const float* fW1  = (const float*)d_in[20];
  const float* fb1  = (const float*)d_in[21];
  const float* fW2  = (const float*)d_in[22];
  const float* fb2  = (const float*)d_in[23];
  const float* fW3  = (const float*)d_in[24];
  const float* fb3  = (const float*)d_in[25];
  const float* fW4  = (const float*)d_in[26];
  const float* fb4  = (const float*)d_in[27];

  half_t* wsH = (half_t*)d_ws;

  // Stage 0: weight f16 swizzle (tiny, L2-resident thereafter)
  wswz_kernel<<<(7 * 8 * 512 + 255) / 256, 256, 0, stream>>>(fW1, 220, 128, 7, 8, wsH + W1S_OFF);
  wswz_kernel<<<(4 * 8 * 512 + 255) / 256, 256, 0, stream>>>(fW2, 128, 128, 4, 8, wsH + W2S_OFF);
  wswz_kernel<<<(4 * 4 * 512 + 255) / 256, 256, 0, stream>>>(fW3, 128, 64, 4, 4, wsH + W3S_OFF);
  wswz_kernel<<<(2 * 1 * 512 + 255) / 256, 256, 0, stream>>>(fW4, 64, 2, 2, 1, wsH + W4S_OFF);

  // Stage 1: GNN -> xcat (f16, B x 224)
  gnn_kernel<<<BB / 64, 64, 0, stream>>>(
      data, dmat, src, dst,
      g0U, g0V, g0A, g0B, g0bU, g0bV, g0bA, g0bB,
      gU, gV, gA, gB, gbU, gbV, gbA, gbB,
      wsH + XCAT_OFF);

  // Stage 2: WMMA MLP -> out (B x 2 f32); 8 waves/block share one LDS
  // weight image (174 KB LDS total of the WGP's 320 KB).
  mlp_kernel<<<BB / 128, 256, 0, stream>>>(wsH, fb1, fb2, fb3, fb4, (float*)d_out);
}
